// CausalSelfAttention_44942537786222
// MI455X (gfx1250) — matrix operand order repaired and checked by reading, not compiled, not run on hardware
//
#include <hip/hip_runtime.h>

// CausalSelfAttention for MI455X (gfx1250, wave32).
// All GEMMs on V_WMMA_F32_16X16X32_F16 (f16 inputs, f32 accumulate): 16384
// FLOP/instr (8x the f32 WMMA), contiguous b128 fragment loads everywhere.
// Flash-style online softmax over 32-key blocks; exp(P) staged in LDS as f16.

#define B_  2
#define T_  2048
#define C_  1024
#define H_  16
#define P_  256
#define HD_ 64

typedef _Float16 v8h  __attribute__((ext_vector_type(8)));
typedef _Float16 v16h __attribute__((ext_vector_type(16)));
typedef float    v8f  __attribute__((ext_vector_type(8)));

// D(16x16 f32) = A(16x32 f16) * B(32x16 f16) + C.
// A lane layout (contiguous, same K order as B): halves [0..7] -> K = hl*16 + j,
// halves [8..15] -> K = hl*16 + 8 + j (hl = lane>>4, m = lane&15).
// B lane layout: col n = lane&15, K = hl*16 + 0..15 contiguous.
// C/D: reg r, lane l -> row m = r + 8*(l>>4), col n = l&15.
__device__ __forceinline__ v8f wmma16(v16h a, v16h b, v8f c) {
  return __builtin_amdgcn_wmma_f32_16x16x32_f16(false, a, false, b,
                                                (short)0, c, false, false);
}
__device__ __forceinline__ v16h cat8(v8h lo, v8h hi) {
  return __builtin_shufflevector(lo, hi, 0, 1, 2, 3, 4, 5, 6, 7,
                                 8, 9, 10, 11, 12, 13, 14, 15);
}

__device__ __forceinline__ float redmax16(float v) {
#pragma unroll
  for (int m = 1; m < 16; m <<= 1) v = fmaxf(v, __shfl_xor(v, m, 32));
  return v;
}
__device__ __forceinline__ float redsum16(float v) {
#pragma unroll
  for (int m = 1; m < 16; m <<= 1) v += __shfl_xor(v, m, 32);
  return v;
}

// ---------------------------------------------------------------------------
// Kernel 0a: elementwise f32 -> f16 convert.
// ---------------------------------------------------------------------------
__global__ void cvt_f16_kernel(const float* __restrict__ src,
                               _Float16* __restrict__ dst, int n) {
  const int i = blockIdx.x * 256 + threadIdx.x;
  if (i < n) dst[i] = (_Float16)src[i];
}

// ---------------------------------------------------------------------------
// Kernel 0b: batched transpose + convert: src (R x C) f32 -> dst (C x R) f16.
// Threads linear over dst (coalesced writes); batch = blockIdx.y.
// ---------------------------------------------------------------------------
__global__ void transpose_cvt_kernel(const float* __restrict__ src,
                                     _Float16* __restrict__ dst,
                                     int R, int C) {
  const size_t batchoff = (size_t)blockIdx.y * R * C;
  const int j = blockIdx.x * 256 + threadIdx.x;  // index into dst
  if (j < R * C) {
    const int c = j / R, r = j - c * R;
    dst[batchoff + j] = (_Float16)src[batchoff + (size_t)r * C + c];
  }
}

// ---------------------------------------------------------------------------
// Kernel 1: f16 projection GEMM + f32 RoPE epilogue + head-major scatter.
// One wave -> 16 rows (of B*T) x 64 cols (one head). z: 0=q, 1=k, 2=v.
// q is pre-scaled by 1/sqrt(HD); v is stored transposed (B,H,HD,T).
// ---------------------------------------------------------------------------
__global__ void proj_rope_kernel(const _Float16* __restrict__ xh,
                                 const _Float16* __restrict__ xkh,
                                 const _Float16* __restrict__ WqT,
                                 const _Float16* __restrict__ WkvT,
                                 const int*      __restrict__ start_ptr,
                                 _Float16* __restrict__ qout,
                                 _Float16* __restrict__ kout,
                                 _Float16* __restrict__ vtout) {
  const int lane = threadIdx.x & 31;
  const int ln = lane & 15, hl = lane >> 4;
  const int mt   = blockIdx.x;   // 16-row tile of M = B*T
  const int h    = blockIdx.y;   // head
  const int what = blockIdx.z;   // 0=q, 1=k, 2=v

  const _Float16* src = (what == 0) ? xh : xkh;
  const _Float16* WT  = (what == 0) ? WqT : WkvT;  // (N x K) row-major
  const int colb = ((what == 2) ? C_ : 0) + h * HD_;

  const int mbase = mt * 16;
  v8f acc[4] = {};

  const _Float16* arow = src + (size_t)(mbase + ln) * C_;
#pragma unroll 2
  for (int ks = 0; ks < C_ / 32; ++ks) {
    const int kb = ks * 32;
    v8h lo = *(const v8h*)(arow + kb + hl * 16);
    v8h hi = *(const v8h*)(arow + kb + hl * 16 + 8);
    v16h a = cat8(lo, hi);
#pragma unroll
    for (int dt = 0; dt < 4; ++dt) {
      v16h b = *(const v16h*)(WT + (size_t)(colb + dt * 16 + ln) * C_ +
                              kb + hl * 16);
      acc[dt] = wmma16(a, b, acc[dt]);
    }
  }

  const int bidx = mbase / T_;
  const int tloc = mbase % T_;
  const int bh   = bidx * H_ + h;
  if (what < 2) {
    // RoPE: dim d (<32) pairs with d+32 => acc[dt] with acc[dt+2], same lane,
    // same register element. Angle index i = (dt&1)*16 + ln.
    const int start = *start_ptr;
#pragma unroll
    for (int pr = 0; pr < 2; ++pr) {
      const float invf =
          __expf(-9.210340371976184f * ((float)(pr * 16 + ln) * (1.0f / 32.0f)));
#pragma unroll
      for (int r = 0; r < 8; ++r) {
        const int t = tloc + r + 8 * hl;
        float sA, cA;
        __sincosf((float)(start + t) * invf, &sA, &cA);
        const float x1 = acc[pr][r], x2 = acc[pr + 2][r];
        acc[pr][r]     = x1 * cA - x2 * sA;
        acc[pr + 2][r] = x1 * sA + x2 * cA;
      }
    }
  }

  if (what == 2) {  // v: store transposed (B,H,HD,T)
    _Float16* dv = vtout + (size_t)bh * HD_ * T_;
#pragma unroll
    for (int dt = 0; dt < 4; ++dt)
#pragma unroll
      for (int r = 0; r < 8; ++r)
        dv[(size_t)(dt * 16 + ln) * T_ + tloc + r + 8 * hl] =
            (_Float16)acc[dt][r];
  } else {
    const float sc = (what == 0) ? 0.125f : 1.0f;  // fold 1/sqrt(HD) into q
    _Float16* drow = ((what == 0) ? qout : kout) + (size_t)bh * T_ * HD_;
#pragma unroll
    for (int dt = 0; dt < 4; ++dt)
#pragma unroll
      for (int r = 0; r < 8; ++r)
        drow[(size_t)(tloc + r + 8 * hl) * HD_ + dt * 16 + ln] =
            (_Float16)(acc[dt][r] * sc);
  }
}

// ---------------------------------------------------------------------------
// Kernel 2: flash attention, 32-key blocks. Vt is transposed (HD x L).
// ---------------------------------------------------------------------------
__device__ __forceinline__ void attn_update32(v8f s0, v8f s1,
                                              const _Float16* __restrict__ Vt,
                                              int ldv, int kb, v8f acc[4],
                                              float rmax[8], float rsum[8],
                                              _Float16* plds, int ln, int hl) {
  float corr[8];
#pragma unroll
  for (int r = 0; r < 8; ++r) {
    const float tm = redmax16(fmaxf(s0[r], s1[r]));
    const float nm = fmaxf(rmax[r], tm);
    const float c  = __expf(rmax[r] - nm);
    const float p0 = __expf(s0[r] - nm);
    const float p1 = __expf(s1[r] - nm);
    rsum[r] = rsum[r] * c + redsum16(p0 + p1);
    rmax[r] = nm;
    corr[r] = c;
    plds[(r + 8 * hl) * 32 + ln]      = (_Float16)p0;   // 16x32 f16 P tile
    plds[(r + 8 * hl) * 32 + 16 + ln] = (_Float16)p1;
  }
#pragma unroll
  for (int dt = 0; dt < 4; ++dt)
#pragma unroll
    for (int r = 0; r < 8; ++r) acc[dt][r] *= corr[r];

  // Read P back as an A-fragment (16B-aligned ds_load_b128; same wave).
  v8h lo = *(const v8h*)(plds + ln * 32 + hl * 16);
  v8h hi = *(const v8h*)(plds + ln * 32 + hl * 16 + 8);
  v16h a = cat8(lo, hi);
#pragma unroll
  for (int dt = 0; dt < 4; ++dt) {
    v16h b = *(const v16h*)(Vt + (size_t)(dt * 16 + ln) * ldv + kb + hl * 16);
    acc[dt] = wmma16(a, b, acc[dt]);
  }
}

__global__ void attn_kernel(const _Float16* __restrict__ q,
                            const _Float16* __restrict__ k,
                            const _Float16* __restrict__ vt,
                            const float*    __restrict__ ap,
                            const _Float16* __restrict__ cvt,
                            _Float16* __restrict__ y) {
  __shared__ __align__(32) _Float16 plds[16 * 32];
  const int lane = threadIdx.x & 31;
  const int ln = lane & 15, hl = lane >> 4;
  const int qt = blockIdx.x;
  const int bh = blockIdx.y;
  const int qbase = qt * 16;

  const _Float16* qp  = q   + (size_t)bh * T_ * HD_;
  const _Float16* kp  = k   + (size_t)bh * T_ * HD_;
  const _Float16* vp  = vt  + (size_t)bh * HD_ * T_;   // (HD x T)
  const float*    app = ap  + (size_t)bh * T_ * P_;
  const _Float16* cvp = cvt + (size_t)bh * HD_ * P_;   // (HD x P)

  // 16x64 query tile as two A-fragments (already scaled by 1/sqrt(HD)).
  v16h qa[2];
#pragma unroll
  for (int ks = 0; ks < 2; ++ks) {
    const _Float16* ar = qp + (size_t)(qbase + ln) * HD_ + ks * 32;
    qa[ks] = cat8(*(const v8h*)(ar + hl * 16), *(const v8h*)(ar + hl * 16 + 8));
  }

  v8f acc[4] = {};
  float rmax[8], rsum[8];
#pragma unroll
  for (int r = 0; r < 8; ++r) { rmax[r] = -1e30f; rsum[r] = 0.0f; }

  // Prefix logits (raw f32, no scaling), V = cache_v^T.
  for (int pt = 0; pt < P_ / 32; ++pt) {
    const int pb = pt * 32;
    v8f s0, s1;
#pragma unroll
    for (int r = 0; r < 8; ++r) {
      const float* arow = app + (size_t)(qbase + r + 8 * hl) * P_ + pb + ln;
      s0[r] = arow[0];
      s1[r] = arow[16];
    }
    attn_update32(s0, s1, cvp, P_, pb, acc, rmax, rsum, plds, ln, hl);
  }

  // Causal key blocks of 32: S via WMMA (k rows load as contiguous v16h).
  const int nb = (qbase + 47) / 32;  // blocks covering keys [0, qbase+16)
  for (int nt = 0; nt < nb; ++nt) {
    const int kb = nt * 32;
    v8f s0 = {}, s1 = {};
#pragma unroll
    for (int ks = 0; ks < 2; ++ks) {
      const int ko = ks * 32 + hl * 16;
      v16h b0 = *(const v16h*)(kp + (size_t)(kb + ln) * HD_ + ko);
      v16h b1 = *(const v16h*)(kp + (size_t)(kb + 16 + ln) * HD_ + ko);
      s0 = wmma16(qa[ks], b0, s0);
      s1 = wmma16(qa[ks], b1, s1);
    }
    if (nt == nb - 1) {  // boundary block: mask key > query
#pragma unroll
      for (int r = 0; r < 8; ++r) {
        const int m = qbase + r + 8 * hl;
        if (kb + ln > m)      s0[r] = -1e30f;
        if (kb + 16 + ln > m) s1[r] = -1e30f;
      }
    }
    attn_update32(s0, s1, vp, T_, kb, acc, rmax, rsum, plds, ln, hl);
  }

  // Normalize, convert to f16, scatter to (B, T, C) for the output GEMM.
  const int bidx = bh / H_, h = bh % H_;
#pragma unroll
  for (int dt = 0; dt < 4; ++dt)
#pragma unroll
    for (int r = 0; r < 8; ++r) {
      const int t = qbase + r + 8 * hl;
      y[(size_t)(bidx * T_ + t) * C_ + h * HD_ + dt * 16 + ln] =
          (_Float16)(acc[dt][r] / rsum[r]);
    }
}

// ---------------------------------------------------------------------------
// Kernel 3: out = y @ Wo (f16 in, f32 out). One wave -> 16x64 tile.
// ---------------------------------------------------------------------------
__global__ void out_gemm_kernel(const _Float16* __restrict__ yh,
                                const _Float16* __restrict__ WoT,
                                float* __restrict__ out) {
  const int lane = threadIdx.x & 31;
  const int ln = lane & 15, hl = lane >> 4;
  const int mbase = blockIdx.x * 16;
  const int nbase = blockIdx.y * 64;

  v8f acc[4] = {};
  const _Float16* arow = yh + (size_t)(mbase + ln) * C_;
#pragma unroll 2
  for (int ks = 0; ks < C_ / 32; ++ks) {
    const int kb = ks * 32;
    v16h a = cat8(*(const v8h*)(arow + kb + hl * 16),
                  *(const v8h*)(arow + kb + hl * 16 + 8));
#pragma unroll
    for (int dt = 0; dt < 4; ++dt) {
      v16h b = *(const v16h*)(WoT + (size_t)(nbase + dt * 16 + ln) * C_ +
                              kb + hl * 16);
      acc[dt] = wmma16(a, b, acc[dt]);
    }
  }
#pragma unroll
  for (int dt = 0; dt < 4; ++dt)
#pragma unroll
    for (int r = 0; r < 8; ++r)
      out[(size_t)(mbase + r + 8 * hl) * C_ + nbase + dt * 16 + ln] =
          acc[dt][r];
}

// ---------------------------------------------------------------------------
extern "C" void kernel_launch(void* const* d_in, const int* in_sizes, int n_in,
                              void* d_out, int out_size, void* d_ws, size_t ws_size,
                              hipStream_t stream) {
  (void)in_sizes; (void)n_in; (void)out_size; (void)ws_size;
  const float* x     = (const float*)d_in[0];
  const float* x_key = (const float*)d_in[1];
  const float* Wq    = (const float*)d_in[2];
  const float* Wkv   = (const float*)d_in[3];
  const float* Wo    = (const float*)d_in[4];
  const float* apf   = (const float*)d_in[5];
  const float* cv    = (const float*)d_in[6];
  const int*   start = (const int*)d_in[7];
  float* out = (float*)d_out;

  const size_t nx   = (size_t)B_ * T_ * C_;        // 4.19M
  const size_t nw   = (size_t)C_ * C_;             // 1.05M
  const size_t ncv  = (size_t)B_ * H_ * HD_ * P_;  // 0.52M
  const size_t nqkv = (size_t)B_ * H_ * T_ * HD_;  // 4.19M

  _Float16* xh   = (_Float16*)d_ws;
  _Float16* xkh  = xh   + nx;
  _Float16* WqT  = xkh  + nx;
  _Float16* WkvT = WqT  + nw;
  _Float16* WoT  = WkvT + 2 * nw;
  _Float16* cvT  = WoT  + nw;
  _Float16* qh   = cvT  + ncv;
  _Float16* kh   = qh   + nqkv;
  _Float16* vT   = kh   + nqkv;
  _Float16* yh   = vT   + nqkv;   // (B, T, C)  total ~60 MB

  // Stage inputs in f16; transpose weights to (N x K) and cache_v to (HD x P)
  // so every WMMA B-fragment is a contiguous 32-byte load.
  cvt_f16_kernel<<<dim3((int)(nx / 256)), 256, 0, stream>>>(x, xh, (int)nx);
  cvt_f16_kernel<<<dim3((int)(nx / 256)), 256, 0, stream>>>(x_key, xkh, (int)nx);
  transpose_cvt_kernel<<<dim3((int)(nw / 256), 1), 256, 0, stream>>>(
      Wq, WqT, C_, C_);
  transpose_cvt_kernel<<<dim3((int)(2 * nw / 256), 1), 256, 0, stream>>>(
      Wkv, WkvT, C_, 2 * C_);
  transpose_cvt_kernel<<<dim3((int)(nw / 256), 1), 256, 0, stream>>>(
      Wo, WoT, C_, C_);
  transpose_cvt_kernel<<<dim3((P_ * HD_) / 256, B_ * H_), 256, 0, stream>>>(
      cv, cvT, P_, HD_);

  proj_rope_kernel<<<dim3((B_ * T_) / 16, H_, 3), 32, 0, stream>>>(
      xh, xkh, WqT, WkvT, start, qh, kh, vT);
  attn_kernel<<<dim3(T_ / 16, B_ * H_), 32, 0, stream>>>(
      qh, kh, vT, apf, cvT, yh);
  out_gemm_kernel<<<dim3((B_ * T_) / 16, C_ / 64), 32, 0, stream>>>(
      yh, WoT, out);
}
